// Transformer_Attention_4466765988310
// MI455X (gfx1250) — matrix operand order repaired and checked by reading, not compiled, not run on hardware
//
#include <hip/hip_runtime.h>
#include <hip/hip_bf16.h>

// ---------- types ----------
typedef __attribute__((ext_vector_type(16))) __bf16       v16bf;
typedef __attribute__((ext_vector_type(8)))  float        v8f;
typedef __attribute__((ext_vector_type(4)))  unsigned int u32x4;
typedef __attribute__((ext_vector_type(4)))  unsigned short u16x4;

#define DEVI __device__ __forceinline__

DEVI unsigned short f2bf(float f) {            // round-to-nearest-even f32->bf16
  unsigned int u = __float_as_uint(f);
  u += 0x7FFFu + ((u >> 16) & 1u);
  return (unsigned short)(u >> 16);
}
DEVI float bf2f(unsigned int h) { return __uint_as_float(h << 16); }

DEVI v8f vzero() { v8f z = {0.f,0.f,0.f,0.f,0.f,0.f,0.f,0.f}; return z; }

// A fragment 16x32 bf16 (ISA 7.12.2): lane = m + 16*half.
// VGPR0-3 hold K = half*8 + 0..7 ; VGPR4-7 hold K = 16 + half*8 + 0..7.
DEVI v16bf frag_a(const unsigned short* base, int stride, int lane) {
  const int m = lane & 15, hf = lane >> 4;
  const unsigned short* p = base + m * stride + hf * 8;
  union { u32x4 q[2]; v16bf v; } U;
  U.q[0] = *(const u32x4*)(p);
  U.q[1] = *(const u32x4*)(p + 16);
  return U.v;
}
// B fragment 32x16 bf16: lane = col + 16*kh; holds K = kh*8 + 0..7 and 16 + kh*8 + 0..7 of column col.
// `base` points at column-major-in-K storage: column c at base + c*stride (K contiguous).
DEVI v16bf frag_b(const unsigned short* base, int stride, int lane) {
  const int c = lane & 15, kh = lane >> 4;
  const unsigned short* p = base + c * stride + kh * 8;
  union { u32x4 q[2]; v16bf v; } U;
  U.q[0] = *(const u32x4*)(p);
  U.q[1] = *(const u32x4*)(p + 16);
  return U.v;
}
DEVI v8f wmma_bf16(v16bf a, v16bf b, v8f c) {
  return __builtin_amdgcn_wmma_f32_16x16x32_bf16(false, a, false, b, (short)0, c, false, false);
}

// ---------- constants ----------
#define CDIM 512
#define NPIX 4096     // 64*64
#define BATCH 8
#define HEADS 8
#define HD 64

// ---------- kernel 1: depthwise 3x3 conv, fp32 -> bf16, transposed store xdT[b][n][c] ----------
__global__ __launch_bounds__(256) void k_dwconv(const float* __restrict__ x,
                                                const float* __restrict__ dww,
                                                unsigned short* __restrict__ xdT) {
  const int tid = threadIdx.x;
  const int b  = blockIdx.z;
  const int hw = blockIdx.x * 64 + (tid & 63);
  const int c0 = blockIdx.y * 32 + (tid >> 6) * 8;
  const int h = hw >> 6, w = hw & 63;
  unsigned int pk[4];
  for (int cc = 0; cc < 8; ++cc) {
    const int c = c0 + cc;
    const float* xp = x   + ((size_t)(b * CDIM + c) << 12);
    const float* wp = dww + c * 9;
    float acc = 0.f;
#pragma unroll
    for (int i = 0; i < 3; ++i) {
      const int hh = h + i - 1;
      if (hh < 0 || hh > 63) continue;
#pragma unroll
      for (int j = 0; j < 3; ++j) {
        const int ww = w + j - 1;
        if (ww < 0 || ww > 63) continue;
        acc += xp[(hh << 6) + ww] * wp[i * 3 + j];
      }
    }
    const unsigned int hb = f2bf(acc);
    if (cc & 1) pk[cc >> 1] |= hb << 16; else pk[cc >> 1] = hb;
  }
  u32x4 o = {pk[0], pk[1], pk[2], pk[3]};
  *(u32x4*)(xdT + ((size_t)b * NPIX + hw) * CDIM + c0) = o;
}

// ---------- kernel 2: fp32 -> bf16 weight convert ----------
__global__ __launch_bounds__(256) void k_cvt(const float* __restrict__ s,
                                             unsigned short* __restrict__ d, int n) {
  const int i = (blockIdx.x * 256 + threadIdx.x) * 4;
  if (i + 3 < n) {
    u16x4 o = { f2bf(s[i]), f2bf(s[i+1]), f2bf(s[i+2]), f2bf(s[i+3]) };
    *(u16x4*)(d + i) = o;
  }
}

// ---------- kernel 3: WMMA GEMM  out[b][o][n] = sum_c W[o][c] * Bt[b][n][c] ----------
// block = 256 thr (8 waves). Wave tile M=64 x N=64 (acc 4x4 v8f = 128 VGPRs):
// 16 b128 loads per 16 WMMAs per k-step (1.0 loads/wmma). Block tile M=64 x N=512.
template<int OUTF32>
__global__ __launch_bounds__(256) void k_gemm(const unsigned short* __restrict__ A,   // [512][512] bf16
                                              const unsigned short* __restrict__ Bt,  // [8][4096][512] bf16
                                              unsigned short* __restrict__ outH,      // [8][512][4096] bf16
                                              float* __restrict__ outF) {             // [8][512][4096] f32
  const int lane = threadIdx.x & 31, wv = threadIdx.x >> 5;
  const int b  = blockIdx.z;
  const int m0 = blockIdx.y * 64;
  const int n0 = blockIdx.x * 512 + wv * 64;
  const unsigned short* Bb = Bt + ((size_t)b * NPIX * CDIM);

  v8f acc[4][4];
#pragma unroll
  for (int mi = 0; mi < 4; ++mi)
#pragma unroll
    for (int ni = 0; ni < 4; ++ni) acc[mi][ni] = vzero();

  for (int kb = 0; kb < CDIM; kb += 32) {
    v16bf af[4], bf[4];
#pragma unroll
    for (int mi = 0; mi < 4; ++mi)
      af[mi] = frag_a(A + (size_t)(m0 + mi * 16) * CDIM + kb, CDIM, lane);
#pragma unroll
    for (int ni = 0; ni < 4; ++ni)
      bf[ni] = frag_b(Bb + (size_t)(n0 + ni * 16) * CDIM + kb, CDIM, lane);
#pragma unroll
    for (int mi = 0; mi < 4; ++mi)
#pragma unroll
      for (int ni = 0; ni < 4; ++ni)
        acc[mi][ni] = wmma_bf16(af[mi], bf[ni], acc[mi][ni]);
  }

  const int ln = lane & 15, lh = lane >> 4;
#pragma unroll
  for (int mi = 0; mi < 4; ++mi)
#pragma unroll
    for (int ni = 0; ni < 4; ++ni) {
      const int m = m0 + mi * 16 + lh * 8;
      const int n = n0 + ni * 16 + ln;
      if (OUTF32) {
        float* ob = outF + ((size_t)b * CDIM) * NPIX;
#pragma unroll
        for (int j = 0; j < 8; ++j) ob[(size_t)(m + j) * NPIX + n] = acc[mi][ni][j];
      } else {
        unsigned short* ob = outH + ((size_t)b * CDIM) * NPIX;
#pragma unroll
        for (int j = 0; j < 8; ++j) ob[(size_t)(m + j) * NPIX + n] = f2bf(acc[mi][ni][j]);
      }
    }
}

// ---------- kernel 4: per-row 1/max(||row||,eps) for q,k,v ----------
__global__ __launch_bounds__(256) void k_rownorm(const unsigned short* __restrict__ q,
                                                 const unsigned short* __restrict__ k,
                                                 const unsigned short* __restrict__ v,
                                                 float* __restrict__ rfac) {
  const int lane = threadIdx.x & 31, wv = threadIdx.x >> 5;
  const int row = blockIdx.x * 8 + wv;
  const int b = blockIdx.y, t = blockIdx.z;
  const unsigned short* p = (t == 0 ? q : t == 1 ? k : v) + ((size_t)(b * CDIM + row) << 12);
  const unsigned int* p32 = (const unsigned int*)p;
  float s = 0.f;
  for (int i = lane; i < 2048; i += 32) {
    const unsigned int u = p32[i];
    const float a = bf2f(u & 0xffffu), c = bf2f(u >> 16);
    s += a * a + c * c;
  }
#pragma unroll
  for (int o = 16; o; o >>= 1) s += __shfl_xor(s, o);
  if (lane == 0) rfac[t * 4096 + b * CDIM + row] = 1.f / fmaxf(sqrtf(s), 1e-12f);
}

// ---------- kernel 5: channel attention for one (b,h): S=q*k^T (WMMA, split-K + ds_add_f32),
// fold norms+scale, softmax, fold 1/|v| into columns, then out = A*v (WMMA via LDS transpose),
// stored transposed attnT[b][n][c] bf16 for the proj GEMM. ----------
__global__ __launch_bounds__(256) void k_attn(const unsigned short* __restrict__ q,
                                              const unsigned short* __restrict__ k,
                                              const unsigned short* __restrict__ v,
                                              const float* __restrict__ rfac,
                                              unsigned short* __restrict__ attnT) {
  __shared__ float S[64][64];                 // 16 KB score accumulator
  __shared__ unsigned short Abf[64][72];      // 9 KB softmaxed A (bf16), pitch 72 halves
  __shared__ unsigned short vT[256][72];      // 36 KB transposed v chunk, pitch 72 halves

  const int tid = threadIdx.x, lane = tid & 31, wv = tid >> 5;
  const int hh = blockIdx.x, b = blockIdx.y;
  const size_t base = ((size_t)(b * CDIM + hh * HD)) << 12;
  const unsigned short* qb = q + base;
  const unsigned short* kb = k + base;
  const unsigned short* vb = v + base;

  for (int i = tid; i < 4096; i += 256) (&S[0][0])[i] = 0.f;
  __syncthreads();

  // ---- phase 1: S += q[64 x Kslice] * k^T, each wave owns K slice of 512 ----
  {
    v8f acc[4][4];
#pragma unroll
    for (int mi = 0; mi < 4; ++mi)
#pragma unroll
      for (int ni = 0; ni < 4; ++ni) acc[mi][ni] = vzero();
    const int k0 = wv * 512;
    for (int kp = k0; kp < k0 + 512; kp += 32) {
      v16bf af[4], bf[4];
#pragma unroll
      for (int mi = 0; mi < 4; ++mi)
        af[mi] = frag_a(qb + (size_t)(mi * 16) * NPIX + kp, NPIX, lane);
#pragma unroll
      for (int ni = 0; ni < 4; ++ni)
        bf[ni] = frag_b(kb + (size_t)(ni * 16) * NPIX + kp, NPIX, lane);
#pragma unroll
      for (int mi = 0; mi < 4; ++mi)
#pragma unroll
        for (int ni = 0; ni < 4; ++ni)
          acc[mi][ni] = wmma_bf16(af[mi], bf[ni], acc[mi][ni]);
    }
    const int ln = lane & 15, lh = lane >> 4;
#pragma unroll
    for (int mi = 0; mi < 4; ++mi)
#pragma unroll
      for (int ni = 0; ni < 4; ++ni)
#pragma unroll
        for (int j = 0; j < 8; ++j)
          atomicAdd(&S[mi * 16 + lh * 8 + j][ni * 16 + ln], acc[mi][ni][j]);  // ds_add_f32
  }
  __syncthreads();

  // ---- phase 1b: fold rq*rk*scale, softmax rows, fold rv into columns, bf16 ----
  {
    const int r = tid >> 2, qd = tid & 3;                 // 4 lanes per row (quad)
    const float* rq = rfac        + b * CDIM + hh * HD;
    const float* rk = rfac + 4096 + b * CDIM + hh * HD;
    const float* rv = rfac + 8192 + b * CDIM + hh * HD;
    const float fr = rq[r] * 0.125f;                      // HEAD_DIM^-0.5
    float xv[16], mx = -1e30f;
#pragma unroll
    for (int i = 0; i < 16; ++i) {
      const int e = qd * 16 + i;
      xv[i] = S[r][e] * (fr * rk[e]);
      mx = fmaxf(mx, xv[i]);
    }
    mx = fmaxf(mx, __shfl_xor(mx, 1));
    mx = fmaxf(mx, __shfl_xor(mx, 2));
    float sm = 0.f;
#pragma unroll
    for (int i = 0; i < 16; ++i) { xv[i] = __expf(xv[i] - mx); sm += xv[i]; }
    sm += __shfl_xor(sm, 1);
    sm += __shfl_xor(sm, 2);
    const float inv = 1.f / sm;
#pragma unroll
    for (int i = 0; i < 16; ++i) {
      const int e = qd * 16 + i;
      Abf[r][e] = f2bf(xv[i] * inv * rv[e]);
    }
  }
  __syncthreads();

  // ---- phase 2: out[64][4096] = A * v, chunks of 256 columns, v transposed into LDS ----
  const int epair = tid & 31, nset = tid >> 5;
  for (int ch = 0; ch < 16; ++ch) {
    const int nb0 = ch * 256;
    { // cooperative transpose: vT[n][e] <- v[e][nb0+n]
      const unsigned short* v0 = vb + (size_t)(2 * epair) * NPIX + nb0 + nset * 32;
      const unsigned short* v1 = v0 + NPIX;
#pragma unroll
      for (int g = 0; g < 4; ++g) {
        union { u32x4 qv; unsigned short s[8]; } A0, A1;
        A0.qv = *(const u32x4*)(v0 + g * 8);
        A1.qv = *(const u32x4*)(v1 + g * 8);
#pragma unroll
        for (int q2 = 0; q2 < 8; ++q2) {
          const int n = nset * 32 + g * 8 + q2;
          ((unsigned int*)(&vT[n][0]))[epair] =
              (unsigned int)A0.s[q2] | ((unsigned int)A1.s[q2] << 16);
        }
      }
    }
    __syncthreads();

    v8f acc2[4][2];
#pragma unroll
    for (int mi = 0; mi < 4; ++mi)
#pragma unroll
      for (int nj = 0; nj < 2; ++nj) acc2[mi][nj] = vzero();
#pragma unroll
    for (int kk = 0; kk < 64; kk += 32) {
      v16bf a2[4], b2[2];
#pragma unroll
      for (int mi = 0; mi < 4; ++mi)
        a2[mi] = frag_a(&Abf[mi * 16][0] + kk, 72, lane);
#pragma unroll
      for (int nj = 0; nj < 2; ++nj)
        b2[nj] = frag_b(&vT[(wv * 2 + nj) * 16][0] + kk, 72, lane);
#pragma unroll
      for (int mi = 0; mi < 4; ++mi)
#pragma unroll
        for (int nj = 0; nj < 2; ++nj)
          acc2[mi][nj] = wmma_bf16(a2[mi], b2[nj], acc2[mi][nj]);
    }
    const int ln = lane & 15, lh = lane >> 4;
#pragma unroll
    for (int mi = 0; mi < 4; ++mi)
#pragma unroll
      for (int nj = 0; nj < 2; ++nj) {
        const int n = nb0 + (wv * 2 + nj) * 16 + ln;
        unsigned int pk[4];
#pragma unroll
        for (int j2 = 0; j2 < 4; ++j2)
          pk[j2] = (unsigned int)f2bf(acc2[mi][nj][2 * j2]) |
                   ((unsigned int)f2bf(acc2[mi][nj][2 * j2 + 1]) << 16);
        u32x4 o = {pk[0], pk[1], pk[2], pk[3]};
        *(u32x4*)(attnT + ((size_t)(b * NPIX + n)) * CDIM + hh * HD + mi * 16 + lh * 8) = o;
      }
    __syncthreads();
  }
}

// ---------- host ----------
extern "C" void kernel_launch(void* const* d_in, const int* in_sizes, int n_in,
                              void* d_out, int out_size, void* d_ws, size_t ws_size,
                              hipStream_t stream) {
  const float* x    = (const float*)d_in[0];
  const float* dw_w = (const float*)d_in[1];
  const float* q_w  = (const float*)d_in[2];
  const float* k_w  = (const float*)d_in[3];
  const float* v_w  = (const float*)d_in[4];
  const float* p_w  = (const float*)d_in[5];

  char* ws = (char*)d_ws;
  const size_t SZ_XDT = (size_t)BATCH * NPIX * CDIM * 2;   // 32 MB (reused as attnT later)
  const size_t SZ_W   = (size_t)CDIM * CDIM * 2;           // 0.5 MB each
  const size_t SZ_QKV = (size_t)BATCH * CDIM * NPIX * 2;   // 32 MB each
  unsigned short* xdT = (unsigned short*)(ws);
  unsigned short* Wq  = (unsigned short*)(ws + SZ_XDT);
  unsigned short* Wk  = (unsigned short*)(ws + SZ_XDT + SZ_W);
  unsigned short* Wv  = (unsigned short*)(ws + SZ_XDT + 2 * SZ_W);
  unsigned short* Wp  = (unsigned short*)(ws + SZ_XDT + 3 * SZ_W);
  unsigned short* qB  = (unsigned short*)(ws + SZ_XDT + 4 * SZ_W);
  unsigned short* kB  = (unsigned short*)((char*)qB + SZ_QKV);
  unsigned short* vB  = (unsigned short*)((char*)kB + SZ_QKV);
  float*          rf  = (float*)((char*)vB + SZ_QKV);      // 3*8*512 f32
  unsigned short* attnT = xdT;                             // reuse: xd dead after v-GEMM

  // 1. depthwise conv -> xdT[b][n][c] bf16
  k_dwconv<<<dim3(64, 16, BATCH), 256, 0, stream>>>(x, dw_w, xdT);
  // 2. weights -> bf16
  k_cvt<<<256, 256, 0, stream>>>(q_w, Wq, CDIM * CDIM);
  k_cvt<<<256, 256, 0, stream>>>(k_w, Wk, CDIM * CDIM);
  k_cvt<<<256, 256, 0, stream>>>(v_w, Wv, CDIM * CDIM);
  k_cvt<<<256, 256, 0, stream>>>(p_w, Wp, CDIM * CDIM);
  // 3. q/k/v 1x1 convs (bf16 WMMA GEMMs), 64x64 wave tiles
  k_gemm<0><<<dim3(8, 8, BATCH), 256, 0, stream>>>(Wq, xdT, qB, nullptr);
  k_gemm<0><<<dim3(8, 8, BATCH), 256, 0, stream>>>(Wk, xdT, kB, nullptr);
  k_gemm<0><<<dim3(8, 8, BATCH), 256, 0, stream>>>(Wv, xdT, vB, nullptr);
  // 4. row norms
  k_rownorm<<<dim3(64, BATCH, 3), 256, 0, stream>>>(qB, kB, vB, rf);
  // 5. channel attention -> attnT[b][n][c] bf16
  k_attn<<<dim3(HEADS, BATCH), 256, 0, stream>>>(qB, kB, vB, rf, attnT);
  // 6. proj 1x1 conv -> d_out f32 [b][c][n]
  k_gemm<1><<<dim3(8, 8, BATCH), 256, 0, stream>>>(Wp, attnT, nullptr, (float*)d_out);
  (void)in_sizes; (void)n_in; (void)out_size; (void)ws_size;
}
